// MoleculeModel_62242666054063
// MI455X (gfx1250) — matrix operand order repaired and checked by reading, not compiled, not run on hardware
//
#include <hip/hip_runtime.h>
#include <hip/hip_bf16.h>

typedef __bf16 bf16;
typedef __attribute__((ext_vector_type(16))) __bf16 v16bf;
typedef __attribute__((ext_vector_type(8)))  __bf16 bf16x8;
typedef __attribute__((ext_vector_type(4)))  __bf16 bf16x4;
typedef __attribute__((ext_vector_type(8)))  float  v8f;

#define NATOMS 60000
#define NBONDS 130000
#define MAXNB  6
#define HDIM   600
#define FATOM  133
#define FBOND  147
#define HP     640   // H padded to multiple of 128
#define FBP    160   // F_BOND padded to multiple of 32
#define AIW    736   // (H + F_ATOM) padded to multiple of 32, layout [a_msg(600)|f_atoms(133)|pad]

// ---------------- GEMM: D = relu((A[M,K] * W^T) + resid + bias), bf16 in, f32 acc, bf16 out ----
// W stored as bf16 [640][lda] rows = original [n][k] (so B(k,n) = W[n][k]); N fixed = 640.
#define BM 128
#define BN 128
#define BK 32
#define LDT 40                     // LDS row stride in elements (80B: bank-spread, 16B aligned)
#define TILE_BYTES (BM * LDT * 2)  // 10240 B per tile buffer

// Async staging: memory -> LDS directly (no VGPR staging), tracked by ASYNCcnt.
__device__ __forceinline__ void stage_async(const bf16* __restrict__ A,
                                            const bf16* __restrict__ W,
                                            unsigned ldsA, unsigned ldsB,
                                            int M, int lda, int m0, int n0, int k0, int tid)
{
    // A tile: 128 rows x 4 chunks of 16B
    #pragma unroll
    for (int i = tid; i < BM * 4; i += 256) {
        int r = i >> 2, c8 = (i & 3) << 3;
        int gm = m0 + r;
        if (gm < M) {
            int      voff = (int)(((size_t)gm * lda + k0 + c8) * 2);   // byte offset, fits in i32
            unsigned l    = ldsA + (unsigned)(r * LDT + c8) * 2;
            asm volatile("global_load_async_to_lds_b128 %0, %1, %2"
                         :: "v"(l), "v"(voff), "s"(A) : "memory");
        }
    }
    // B tile: 128 rows x 4 chunks of 16B (no bounds: W padded to 640 rows)
    #pragma unroll
    for (int i = tid; i < BN * 4; i += 256) {
        int r = i >> 2, c8 = (i & 3) << 3;
        int      voff = (int)(((size_t)(n0 + r) * lda + k0 + c8) * 2);
        unsigned l    = ldsB + (unsigned)(r * LDT + c8) * 2;
        asm volatile("global_load_async_to_lds_b128 %0, %1, %2"
                     :: "v"(l), "v"(voff), "s"(W) : "memory");
    }
}

__device__ __forceinline__ void wait_async0()
{
    asm volatile("s_wait_asynccnt 0x0" ::: "memory");
}

__global__ __launch_bounds__(256)
void k_gemm(const bf16* __restrict__ A, const bf16* __restrict__ W,
            const bf16* __restrict__ resid, const float* __restrict__ bias,
            bf16* __restrict__ D, bf16* __restrict__ Dpre,
            int M, int lda, int KT)
{
    __shared__ bf16 lsA[2][BM][LDT];
    __shared__ bf16 lsB[2][BN][LDT];
    const int tid  = threadIdx.x;
    const int m0   = blockIdx.x * BM;
    const int n0   = blockIdx.y * BN;
    const int wave = tid >> 5, lane = tid & 31;
    const int wm   = wave & 3;         // 4 waves along M -> 32 rows each
    const int wn   = wave >> 2;        // 2 waves along N -> 64 cols each
    const int half = lane >> 4, lm = lane & 15;

    const unsigned ldsA0 = (unsigned)(uintptr_t)&lsA[0][0][0];
    const unsigned ldsB0 = (unsigned)(uintptr_t)&lsB[0][0][0];

    // Partial last M-block: pre-zero OOB LDS rows (both buffers). Async loads never
    // touch these rows, so the zeros persist across all K-steps.
    if (m0 + BM > M) {
        for (int i = tid; i < BM * 4; i += 256) {
            int r = i >> 2, c8 = (i & 3) << 3;
            if (m0 + r >= M) {
                bf16x8 z;
                #pragma unroll
                for (int q = 0; q < 8; ++q) z[q] = (bf16)0.0f;
                *(bf16x8*)(&lsA[0][r][c8]) = z;
                *(bf16x8*)(&lsA[1][r][c8]) = z;
            }
        }
    }

    v8f acc[2][4];
    #pragma unroll
    for (int i = 0; i < 2; ++i)
        #pragma unroll
        for (int j = 0; j < 4; ++j)
            #pragma unroll
            for (int r = 0; r < 8; ++r) acc[i][j][r] = 0.0f;

    stage_async(A, W, ldsA0, ldsB0, M, lda, m0, n0, 0, tid);
    wait_async0();
    __syncthreads();

    int buf = 0;
    for (int kt = 0; kt < KT; ++kt) {
        if (kt + 1 < KT)
            stage_async(A, W, ldsA0 + (buf ^ 1) * TILE_BYTES, ldsB0 + (buf ^ 1) * TILE_BYTES,
                        M, lda, m0, n0, (kt + 1) * BK, tid);

        // A fragment (16-bit A 16x32): lanes 0-15 hold K 0-7/16-23, lanes 16-31 hold K 8-15/24-31
        v16bf afr[2], bfr[4];
        #pragma unroll
        for (int mt = 0; mt < 2; ++mt) {
            int r = wm * 32 + mt * 16 + lm;
            bf16x8 lo = *(const bf16x8*)(&lsA[buf][r][8 * half]);
            bf16x8 hi = *(const bf16x8*)(&lsA[buf][r][16 + 8 * half]);
            #pragma unroll
            for (int q = 0; q < 8; ++q) { afr[mt][q] = lo[q]; afr[mt][q + 8] = hi[q]; }
        }
        // B fragment (B 32x16): lane = N, lanes 0-15 K 0-7/16-23, lanes 16-31 K 8-15/24-31
        #pragma unroll
        for (int nt = 0; nt < 4; ++nt) {
            int r = wn * 64 + nt * 16 + lm;
            bf16x8 lo = *(const bf16x8*)(&lsB[buf][r][8 * half]);
            bf16x8 hi = *(const bf16x8*)(&lsB[buf][r][16 + 8 * half]);
            #pragma unroll
            for (int q = 0; q < 8; ++q) { bfr[nt][q] = lo[q]; bfr[nt][q + 8] = hi[q]; }
        }
        #pragma unroll
        for (int mt = 0; mt < 2; ++mt)
            #pragma unroll
            for (int nt = 0; nt < 4; ++nt)
                acc[mt][nt] = __builtin_amdgcn_wmma_f32_16x16x32_bf16(
                    false, afr[mt], false, bfr[nt], (short)0, acc[mt][nt], false, false);

        wait_async0();       // our wave's async loads into buf^1 are done
        __syncthreads();     // everyone's are; safe to read next iter / overwrite this buf
        buf ^= 1;
    }

    // epilogue: C/D layout -> lane n = lm, vgpr r -> row r + 8*half
    #pragma unroll
    for (int mt = 0; mt < 2; ++mt) {
        #pragma unroll
        for (int rr = 0; rr < 8; ++rr) {
            int gm = m0 + wm * 32 + mt * 16 + rr + 8 * half;
            if (gm < M) {
                #pragma unroll
                for (int nt = 0; nt < 4; ++nt) {
                    int gn = n0 + wn * 64 + nt * 16 + lm;
                    float v = acc[mt][nt][rr];
                    if (bias)  v += bias[gn];
                    if (resid) v += (float)resid[(size_t)gm * HP + gn];
                    if (Dpre)  Dpre[(size_t)gm * HP + gn] = (bf16)v;
                    v = v > 0.0f ? v : 0.0f;
                    D[(size_t)gm * HP + gn] = (bf16)v;
                }
            }
        }
    }
}

// ---------------- weight / feature conversion kernels ----------------------------------------
__global__ void k_cvt_w(const float* __restrict__ src, bf16* __restrict__ dst,
                        int Ksrc, int Kpad, int mode, int total)
{
    int gid = blockIdx.x * blockDim.x + threadIdx.x;
    if (gid >= total) return;
    int n = gid / Kpad, k = gid % Kpad;
    float v = 0.0f;
    if (n < HDIM) {
        if (mode == 0) {
            if (k < Ksrc) v = src[(size_t)n * Ksrc + k];
        } else { // W_o, remapped for a_input layout [a_msg | f_atoms]
            if (k < HDIM)      v = src[(size_t)n * 733 + (FATOM + k)];
            else if (k < 733)  v = src[(size_t)n * 733 + (k - HDIM)];
        }
    }
    dst[(size_t)n * Kpad + k] = (bf16)v;
}

__global__ void k_bias_pad(const float* __restrict__ src, float* __restrict__ dst)
{
    int gid = blockIdx.x * blockDim.x + threadIdx.x;
    if (gid >= HP) return;
    dst[gid] = (gid < HDIM) ? src[gid] : 0.0f;
}

__global__ void k_cvt_bonds(const float* __restrict__ fb, bf16* __restrict__ out)
{
    int gid = blockIdx.x * blockDim.x + threadIdx.x;
    if (gid >= NBONDS * 40) return;
    int row = gid / 40, c4 = (gid % 40) * 4;
    bf16x4 o;
    #pragma unroll
    for (int i = 0; i < 4; ++i) {
        int c = c4 + i;
        float v = (c < FBOND) ? fb[(size_t)row * FBOND + c] : 0.0f;
        o[i] = (bf16)v;
    }
    *(bf16x4*)(out + (size_t)row * FBP + c4) = o;
}

__global__ void k_fill_atoms(const float* __restrict__ fa, bf16* __restrict__ a_input)
{
    int gid = blockIdx.x * blockDim.x + threadIdx.x;
    if (gid >= NATOMS * 136) return;
    int a = gid / 136, c = gid % 136;          // cols 600..735 of a_input
    float v = (c < FATOM) ? fa[(size_t)a * FATOM + c] : 0.0f;
    a_input[(size_t)a * AIW + HDIM + c] = (bf16)v;
}

// ---------------- message-passing gather kernels ---------------------------------------------
__global__ void k_atom_sum(const bf16* __restrict__ msg, const int* __restrict__ a2b,
                           bf16* __restrict__ outp, int ostride, int nch)
{
    int gid = blockIdx.x * blockDim.x + threadIdx.x;
    if (gid >= NATOMS * nch) return;
    int a = gid / nch, c4 = (gid % nch) * 4;
    float acc[4] = {0.f, 0.f, 0.f, 0.f};
    #pragma unroll
    for (int j = 0; j < MAXNB; ++j) {
        int b = a2b[a * MAXNB + j];
        bf16x4 v = *(const bf16x4*)(msg + (size_t)b * HP + c4);
        #pragma unroll
        for (int i = 0; i < 4; ++i) acc[i] += (float)v[i];
    }
    bf16x4 o;
    #pragma unroll
    for (int i = 0; i < 4; ++i) o[i] = (bf16)acc[i];
    *(bf16x4*)(outp + (size_t)a * ostride + c4) = o;
}

__global__ void k_bond_pre(const bf16* __restrict__ amsg, const bf16* __restrict__ msg,
                           const int* __restrict__ b2a, const int* __restrict__ b2revb,
                           bf16* __restrict__ outp)
{
    int gid = blockIdx.x * blockDim.x + threadIdx.x;
    if (gid >= NBONDS * 160) return;
    int b = gid / 160, c4 = (gid % 160) * 4;
    int sa = b2a[b], rb = b2revb[b];
    bf16x4 va = *(const bf16x4*)(amsg + (size_t)sa * HP + c4);
    bf16x4 vm = *(const bf16x4*)(msg  + (size_t)rb * HP + c4);
    bf16x4 o;
    #pragma unroll
    for (int i = 0; i < 4; ++i) o[i] = (bf16)((float)va[i] - (float)vm[i]);
    *(bf16x4*)(outp + (size_t)b * HP + c4) = o;
}

// ---------------- readout: sigmoid(h2 . W_r + b_r), drop atom 0 ------------------------------
__global__ void k_readout(const bf16* __restrict__ h2, const float* __restrict__ Wr,
                          const float* __restrict__ br, float* __restrict__ out)
{
    int wave = threadIdx.x >> 5, lane = threadIdx.x & 31;
    int a = blockIdx.x * 8 + wave;
    if (a >= NATOMS) return;
    float s = 0.0f;
    for (int h = lane; h < HDIM; h += 32)
        s += (float)h2[(size_t)a * HP + h] * Wr[h];
    #pragma unroll
    for (int off = 16; off > 0; off >>= 1) s += __shfl_down(s, off, 32);
    if (lane == 0 && a >= 1)
        out[a - 1] = 1.0f / (1.0f + __expf(-(s + br[0])));
}

// ---------------- host orchestration ----------------------------------------------------------
extern "C" void kernel_launch(void* const* d_in, const int* in_sizes, int n_in,
                              void* d_out, int out_size, void* d_ws, size_t ws_size,
                              hipStream_t stream)
{
    const float* f_atoms = (const float*)d_in[0];
    const float* f_bonds = (const float*)d_in[1];
    const int*   a2b     = (const int*)d_in[2];
    const int*   b2a     = (const int*)d_in[3];
    const int*   b2revb  = (const int*)d_in[4];
    const float* W_i     = (const float*)d_in[5];
    const float* W_h     = (const float*)d_in[6];
    const float* W_o     = (const float*)d_in[7];
    const float* b_o     = (const float*)d_in[8];
    const float* W_f1    = (const float*)d_in[9];
    const float* b_f1    = (const float*)d_in[10];
    const float* W_f2    = (const float*)d_in[11];
    const float* b_f2    = (const float*)d_in[12];
    const float* W_r     = (const float*)d_in[13];
    const float* b_r     = (const float*)d_in[14];
    float* out = (float*)d_out;

    // carve workspace
    char* p = (char*)d_ws;
    size_t used = 0;
    auto carve = [&](size_t bytes) -> char* {
        char* q = p;
        size_t pad = (bytes + 255) & ~(size_t)255;
        p += pad; used += pad;
        return q;
    };
    bf16*  Wb_i   = (bf16*)carve((size_t)HP * FBP * 2);
    bf16*  Wb_h   = (bf16*)carve((size_t)HP * HP * 2);
    bf16*  Wb_o   = (bf16*)carve((size_t)HP * AIW * 2);
    bf16*  Wb_f1  = (bf16*)carve((size_t)HP * HP * 2);
    bf16*  Wb_f2  = (bf16*)carve((size_t)HP * HP * 2);
    float* bO     = (float*)carve((size_t)HP * 4);
    float* bF1    = (float*)carve((size_t)HP * 4);
    float* bF2    = (float*)carve((size_t)HP * 4);
    bf16*  Abonds = (bf16*)carve((size_t)NBONDS * FBP * 2);
    bf16*  inp    = (bf16*)carve((size_t)NBONDS * HP * 2);
    bf16*  msgA   = (bf16*)carve((size_t)NBONDS * HP * 2);
    bf16*  msgB   = (bf16*)carve((size_t)NBONDS * HP * 2);
    bf16*  amsg   = (bf16*)carve((size_t)NATOMS * HP * 2);
    bf16*  ainput = (bf16*)carve((size_t)NATOMS * AIW * 2);
    bf16*  atomsA = (bf16*)carve((size_t)NATOMS * HP * 2);
    bf16*  atomsB = (bf16*)carve((size_t)NATOMS * HP * 2);
    if (used > ws_size) return; // workspace too small; bail deterministically

    const int TB = 256;
    auto cdiv = [](long long a, long long b) { return (int)((a + b - 1) / b); };

    // weight / feature conversion (independent)
    k_cvt_w<<<cdiv((long long)HP * FBP, TB), TB, 0, stream>>>(W_i,  Wb_i,  FBOND, FBP, 0, HP * FBP);
    k_cvt_w<<<cdiv((long long)HP * HP,  TB), TB, 0, stream>>>(W_h,  Wb_h,  HDIM,  HP,  0, HP * HP);
    k_cvt_w<<<cdiv((long long)HP * AIW, TB), TB, 0, stream>>>(W_o,  Wb_o,  733,   AIW, 1, HP * AIW);
    k_cvt_w<<<cdiv((long long)HP * HP,  TB), TB, 0, stream>>>(W_f1, Wb_f1, HDIM,  HP,  0, HP * HP);
    k_cvt_w<<<cdiv((long long)HP * HP,  TB), TB, 0, stream>>>(W_f2, Wb_f2, HDIM,  HP,  0, HP * HP);
    k_bias_pad<<<cdiv(HP, TB), TB, 0, stream>>>(b_o,  bO);
    k_bias_pad<<<cdiv(HP, TB), TB, 0, stream>>>(b_f1, bF1);
    k_bias_pad<<<cdiv(HP, TB), TB, 0, stream>>>(b_f2, bF2);
    k_cvt_bonds<<<cdiv((long long)NBONDS * 40, TB), TB, 0, stream>>>(f_bonds, Abonds);
    k_fill_atoms<<<cdiv((long long)NATOMS * 136, TB), TB, 0, stream>>>(f_atoms, ainput);

    const int MTb = cdiv(NBONDS, BM);   // 1016 blocks over bonds
    const int MTa = cdiv(NATOMS, BM);   // 469 blocks over atoms
    dim3 gb(MTb, HP / BN), ga(MTa, HP / BN);

    // inp = f_bonds @ W_i.T (stored pre-relu); msgA = relu(inp)
    k_gemm<<<gb, TB, 0, stream>>>(Abonds, Wb_i, nullptr, nullptr, msgA, inp,
                                  NBONDS, FBP, FBP / BK);

    // 5 message-passing iterations
    for (int it = 0; it < 5; ++it) {
        k_atom_sum<<<cdiv((long long)NATOMS * 160, TB), TB, 0, stream>>>(msgA, a2b, amsg, HP, 160);
        k_bond_pre<<<cdiv((long long)NBONDS * 160, TB), TB, 0, stream>>>(amsg, msgA, b2a, b2revb, msgB);
        k_gemm<<<gb, TB, 0, stream>>>(msgB, Wb_h, inp, nullptr, msgA, nullptr,
                                      NBONDS, HP, HP / BK);
    }

    // final atom aggregation into a_input cols [0,600); f_atoms already in cols [600,733)
    k_atom_sum<<<cdiv((long long)NATOMS * 150, TB), TB, 0, stream>>>(msgA, a2b, ainput, AIW, 150);

    // atom_hiddens = relu(a_input @ W_o.T + b_o)
    k_gemm<<<ga, TB, 0, stream>>>(ainput, Wb_o, nullptr, bO, atomsA, nullptr,
                                  NATOMS, AIW, AIW / BK);
    // FFN
    k_gemm<<<ga, TB, 0, stream>>>(atomsA, Wb_f1, nullptr, bF1, atomsB, nullptr,
                                  NATOMS, HP, HP / BK);
    k_gemm<<<ga, TB, 0, stream>>>(atomsB, Wb_f2, nullptr, bF2, atomsA, nullptr,
                                  NATOMS, HP, HP / BK);
    // readout
    k_readout<<<cdiv(NATOMS, 8), TB, 0, stream>>>(atomsA, W_r, b_r, out);
    (void)in_sizes; (void)n_in; (void)out_size;
}
